// RNNDecoder_34548716929268
// MI455X (gfx1250) — matrix operand order repaired and checked by reading, not compiled, not run on hardware
//
#include <hip/hip_runtime.h>
#include <hip/hip_bf16.h>
#include <stdint.h>

typedef unsigned short u16t;
typedef __attribute__((ext_vector_type(16))) __bf16 v16bf;
typedef __attribute__((ext_vector_type(8)))  float  v8f;
typedef __attribute__((ext_vector_type(4)))  unsigned int v4u;
typedef __attribute__((ext_vector_type(8)))  int v8i;
typedef __attribute__((ext_vector_type(4)))  int v4i;

#define BB     4096
#define HH     512
#define DD     512
#define VV     128
#define MAXLEN 100

#if defined(__HIP_DEVICE_COMPILE__) && __has_builtin(__builtin_amdgcn_tensor_load_to_lds)
#define HAVE_TDM 1
#else
#define HAVE_TDM 0
#endif

static __device__ __forceinline__ u16t f2bf(float f) {
  union { float f; unsigned u; } v; v.f = f;
  unsigned r = v.u + 0x7FFFu + ((v.u >> 16) & 1u);   // round-to-nearest-even
  return (u16t)(r >> 16);
}

union AB16 { uint4 q[2]; v16bf v; };

static __device__ __forceinline__ v8f zero8() {
  v8f z;
#pragma unroll
  for (int i = 0; i < 8; ++i) z[i] = 0.0f;
  return z;
}

static __device__ __forceinline__ v8f wmma_bf16(const AB16& a, const AB16& b, v8f c) {
  return __builtin_amdgcn_wmma_f32_16x16x32_bf16(false, a.v, false, b.v,
                                                 (short)0, c, false, false);
}

// ---------------------------------------------------------------------------
// Generic fp32 -> bf16 convert
// ---------------------------------------------------------------------------
__global__ void k_f2bf(const float* __restrict__ src, u16t* __restrict__ dst, int n) {
  int i = blockIdx.x * blockDim.x + threadIdx.x;
  if (i < n) dst[i] = f2bf(src[i]);
}

// ---------------------------------------------------------------------------
// Pack GRU weights: W' [2048][Kx+512] bf16, gate-interleaved in groups of 16.
// packed col c: g=c>>6, gate=(c&63)>>4, jl=c&15, j=g*16+jl
//   gate0 (r):   [Wih_r  | Whh_r ]      gate1 (z):   [Wih_z  | Whh_z ]
//   gate2 (inn): [Wih_n  |   0   ]      gate3 (hn):  [  0    | Whh_n ]
// ---------------------------------------------------------------------------
__global__ void k_pack_gates(const float* __restrict__ Wih, const float* __restrict__ Whh,
                             const float* __restrict__ bih, const float* __restrict__ bhh,
                             u16t* __restrict__ Wp, float* __restrict__ bp, int Kx) {
  int c = blockIdx.x;                    // 0..2047
  int g = c >> 6, rem = c & 63, gate = rem >> 4, jl = rem & 15;
  int j = g * 16 + jl;
  int K = Kx + HH;
  int row = (gate == 0) ? j : (gate == 1) ? (HH + j) : (2 * HH + j);
  for (int k = threadIdx.x; k < K; k += blockDim.x) {
    bool inx = k < Kx;
    int  kk  = inx ? k : k - Kx;
    float v;
    if (gate <= 1)      v = inx ? Wih[(size_t)row * Kx + kk] : Whh[(size_t)row * HH + kk];
    else if (gate == 2) v = inx ? Wih[(size_t)row * Kx + kk] : 0.0f;
    else                v = inx ? 0.0f : Whh[(size_t)row * HH + kk];
    Wp[(size_t)c * K + k] = f2bf(v);
  }
  if (threadIdx.x == 0) {
    float b;
    if (gate == 0 || gate == 1) b = bih[row] + bhh[row];
    else if (gate == 2)         b = bih[row];
    else                        b = bhh[row];
    bp[c] = b;
  }
}

// ---------------------------------------------------------------------------
// Sequence state init
// ---------------------------------------------------------------------------
__global__ void k_init_seq(int* __restrict__ tok, int* __restrict__ eos,
                           int* __restrict__ x_out, int* __restrict__ end_pads) {
  int b = blockIdx.x * blockDim.x + threadIdx.x;
  if (b < BB) {
    tok[b] = 1;
    eos[b] = 0;
    x_out[(size_t)b * MAXLEN] = 1;
    end_pads[b] = MAXLEN;
  }
}

// ---------------------------------------------------------------------------
// Fused WMMA GEMM + GRU gate epilogue (mode 0) or plain bias epilogue (mode 1).
// Output tile 128x64, 8 waves (4 row x 2 col), 2x2 16x16 WMMA tiles per wave.
// A: up to 3 segments of 512 bf16 (seg0 may be emb[tok[b]]), loaded from global.
// B: weight rows; staged per 256-K chunk into LDS by the Tensor Data Mover
//    (tensor_load_to_lds + s_wait_tensorcnt), with D# padding for a 272-half
//    row stride (bank-conflict-free ds_load_b128 fragment reads).
// ---------------------------------------------------------------------------
__global__ __launch_bounds__(256)
void k_gru_gemm(const u16t* __restrict__ s0, const u16t* __restrict__ s1,
                const u16t* __restrict__ s2,
                const int* __restrict__ tok, const u16t* __restrict__ emb,
                const u16t* __restrict__ Wp, int K,
                const float* __restrict__ bias,
                const float* __restrict__ h_old_f,
                float* __restrict__ hf_out, u16t* __restrict__ hb_out,
                float* __restrict__ hf_out2, u16t* __restrict__ hb_out2,
                int mode) {
  __shared__ __align__(16) char smem[128 * 72 * 4];   // 36864B: B-stage / Ot union

  const int tid  = threadIdx.x;
  const int wv   = tid >> 5;
  const int lane = tid & 31;
  const int wr   = wv & 3;          // wave row 0..3
  const int wc   = wv >> 2;         // wave col 0..1
  const int ml   = lane & 15;
  const int hi   = lane >> 4;       // half-wave select
  const int bm   = blockIdx.y * 128;
  const int c0   = blockIdx.x * 64;

  const int m0 = bm + 32 * wr + ml;     // A rows, fragment i=0
  const int m1 = m0 + 16;               // fragment i=1

  v8f acc00 = zero8(), acc01 = zero8(), acc10 = zero8(), acc11 = zero8();

#if !HAVE_TDM
  const u16t* pB0 = Wp + (size_t)(c0 + 32 * wc + ml) * K;
  const u16t* pB1 = pB0 + (size_t)16 * K;
#endif

  const int ns = K >> 9;                // number of 512-wide K segments
  int kkg = 0;
  for (int s = 0; s < ns; ++s) {
    const u16t* pa0;
    const u16t* pa1;
    if (s == 0 && tok) {                // embedding gather segment
      pa0 = emb + (size_t)tok[m0] * HH;
      pa1 = emb + (size_t)tok[m1] * HH;
    } else {
      const u16t* segp = (s == 0) ? s0 : (s == 1) ? s1 : s2;
      pa0 = segp + (size_t)m0 * 512;
      pa1 = segp + (size_t)m1 * 512;
    }
#if HAVE_TDM
    for (int half = 0; half < 2; ++half) {
      __syncthreads();                  // prior chunk fully consumed
      if (wv == 0) {
        // D# descriptor: 2D tile 256(K) x 64(rows) of 2B elems, row-major,
        // LDS padding 8 DWORDs every 128 DWORDs -> 272-half LDS row stride.
        unsigned long long ga =
            (unsigned long long)(uintptr_t)(Wp + (size_t)c0 * K + kkg + half * 256);
        unsigned lds0 = (unsigned)(uintptr_t)(void*)smem;
        v4u g0;
        g0[0] = 1u;                                   // count=1
        g0[1] = lds0;                                 // lds_addr
        g0[2] = (unsigned)ga;                         // global_addr lo
        g0[3] = ((unsigned)(ga >> 32) & 0x01FFFFFFu) | (2u << 30);  // hi | type=2
        v8i g1;
        g1[0] = (int)((1u << 16) | (1u << 20) | (6u << 22) | (7u << 25));
        //          data_size=2B | pad_en     | interval=128DW | amount=8DW
        g1[1] = (int)(((unsigned)K & 0xFFFFu) << 16);               // tensor_dim0 lo
        g1[2] = (int)((((unsigned)K >> 16) & 0xFFFFu) | (2048u << 16)); // dim0 hi|dim1 lo
        g1[3] = (int)(256u << 16);                    // dim1 hi=0 | tile_dim0=256
        g1[4] = 64;                                   // tile_dim1=64
        g1[5] = K;                                    // tensor_dim0_stride lo
        g1[6] = 0;
        g1[7] = 0;
        v4i gz; gz[0] = 0; gz[1] = 0; gz[2] = 0; gz[3] = 0;
#if defined(__clang_major__) && __clang_major__ >= 23
        v8i gz8; gz8[0]=0; gz8[1]=0; gz8[2]=0; gz8[3]=0; gz8[4]=0; gz8[5]=0; gz8[6]=0; gz8[7]=0;
        __builtin_amdgcn_tensor_load_to_lds(g0, g1, gz, gz, gz8, 0);
#else
        __builtin_amdgcn_tensor_load_to_lds(g0, g1, gz, gz, 0);
#endif
#if __has_builtin(__builtin_amdgcn_s_wait_tensorcnt)
        __builtin_amdgcn_s_wait_tensorcnt(0);         // TENSORcnt is per-wave
#endif
      }
      __syncthreads();                  // chunk visible to all waves
      const u16t (*Bs)[272] = (const u16t (*)[272])smem;
      const int nb = 32 * wc + ml;
#pragma unroll 4
      for (int kl = 0; kl < 256; kl += 32) {
        AB16 a0, a1, b0, b1;
        const int ka = half * 256 + kl + hi * 8;
        a0.q[0] = *(const uint4*)(pa0 + ka);
        a0.q[1] = *(const uint4*)(pa0 + ka + 16);
        a1.q[0] = *(const uint4*)(pa1 + ka);
        a1.q[1] = *(const uint4*)(pa1 + ka + 16);
        __builtin_prefetch(pa0 + ka + 64, 0, 1);      // global_prefetch_b8
        const int kb = kl + hi * 8;
        b0.q[0] = *(const uint4*)(&Bs[nb][kb]);
        b0.q[1] = *(const uint4*)(&Bs[nb][kb + 16]);
        b1.q[0] = *(const uint4*)(&Bs[nb + 16][kb]);
        b1.q[1] = *(const uint4*)(&Bs[nb + 16][kb + 16]);
        acc00 = wmma_bf16(a0, b0, acc00);
        acc01 = wmma_bf16(a0, b1, acc01);
        acc10 = wmma_bf16(a1, b0, acc10);
        acc11 = wmma_bf16(a1, b1, acc11);
      }
    }
#else
#pragma unroll 4
    for (int kl = 0; kl < 512; kl += 32) {
      AB16 a0, a1, b0, b1;
      const int ka = kl + hi * 8;
      a0.q[0] = *(const uint4*)(pa0 + ka);
      a0.q[1] = *(const uint4*)(pa0 + ka + 16);
      a1.q[0] = *(const uint4*)(pa1 + ka);
      a1.q[1] = *(const uint4*)(pa1 + ka + 16);
      const int kb = kkg + kl + hi * 8;
      b0.q[0] = *(const uint4*)(pB0 + kb);
      b0.q[1] = *(const uint4*)(pB0 + kb + 16);
      b1.q[0] = *(const uint4*)(pB1 + kb);
      b1.q[1] = *(const uint4*)(pB1 + kb + 16);
      __builtin_prefetch(pB0 + kb + 128, 0, 1);
      acc00 = wmma_bf16(a0, b0, acc00);
      acc01 = wmma_bf16(a0, b1, acc01);
      acc10 = wmma_bf16(a1, b0, acc10);
      acc11 = wmma_bf16(a1, b1, acc11);
    }
#endif
    kkg += 512;
  }

  __syncthreads();                      // B-stage buffer dead; reuse as Ot
  float (*Ot)[72] = (float (*)[72])smem;
  const int orow = 32 * wr + hi * 8;
  const int ocol = 32 * wc + ml;
#pragma unroll
  for (int r = 0; r < 8; ++r) {
    Ot[orow + r][ocol]           = acc00[r];
    Ot[orow + r][ocol + 16]      = acc01[r];
    Ot[orow + 16 + r][ocol]      = acc10[r];
    Ot[orow + 16 + r][ocol + 16] = acc11[r];
  }
  __syncthreads();

  if (mode == 0) {
    // GRU gate epilogue: 128 rows x 16 hidden units; j0 = c0/4
#pragma unroll
    for (int r = 0; r < 8; ++r) {
      int idx = tid + r * 256;
      int bl = idx >> 4, jl = idx & 15;
      float rv = Ot[bl][jl]      + bias[c0 + jl];
      float zv = Ot[bl][16 + jl] + bias[c0 + 16 + jl];
      float iv = Ot[bl][32 + jl] + bias[c0 + 32 + jl];
      float hv = Ot[bl][48 + jl] + bias[c0 + 48 + jl];
      float rg = 1.0f / (1.0f + __expf(-rv));
      float zg = 1.0f / (1.0f + __expf(-zv));
      float ng = tanhf(iv + rg * hv);
      int j = (c0 >> 2) + jl;
      size_t off = (size_t)(bm + bl) * HH + j;
      float ho = h_old_f[off];
      float hn = (1.0f - zg) * ng + zg * ho;
      hf_out[off] = hn;
      hb_out[off] = f2bf(hn);
    }
  } else {
    // init: h0 = z @ lat_W^T + lat_b, broadcast to both layers
#pragma unroll
    for (int r = 0; r < 32; ++r) {
      int idx = tid + r * 256;
      int bl = idx >> 6, cl = idx & 63;
      float v = Ot[bl][cl] + bias[c0 + cl];
      size_t off = (size_t)(bm + bl) * HH + c0 + cl;
      u16t vb = f2bf(v);
      hf_out[off]  = v;  hb_out[off]  = vb;
      hf_out2[off] = v;  hb_out2[off] = vb;
    }
  }
}

// ---------------------------------------------------------------------------
// logits = h1 @ out_W^T + out_b, argmax, EOS bookkeeping, token emit.
// ---------------------------------------------------------------------------
__global__ __launch_bounds__(256)
void k_logits_argmax(const u16t* __restrict__ hb, const u16t* __restrict__ owb,
                     const float* __restrict__ out_b,
                     int* __restrict__ tok, int* __restrict__ eos,
                     int* __restrict__ x_out, int* __restrict__ end_pads, int t) {
  __shared__ float Lt[16][136];
  const int tid  = threadIdx.x;
  const int wv   = tid >> 5;
  const int lane = tid & 31;
  const int ml   = lane & 15;
  const int hi   = lane >> 4;
  const int bm   = blockIdx.x * 16;

  const u16t* pa = hb  + (size_t)(bm + ml) * HH;
  const u16t* pb = owb + (size_t)(wv * 16 + ml) * HH;

  v8f acc = zero8();
#pragma unroll 4
  for (int kk = 0; kk < 512; kk += 32) {
    AB16 a, b;
    const int ka = kk + hi * 8;
    a.q[0] = *(const uint4*)(pa + ka);
    a.q[1] = *(const uint4*)(pa + ka + 16);
    const int kb = kk + hi * 8;
    b.q[0] = *(const uint4*)(pb + kb);
    b.q[1] = *(const uint4*)(pb + kb + 16);
    acc = wmma_bf16(a, b, acc);
  }
#pragma unroll
  for (int r = 0; r < 8; ++r) Lt[hi * 8 + r][wv * 16 + ml] = acc[r];
  __syncthreads();

  if (tid < 16) {
    int b = bm + tid;
    float best = -3.4e38f;
    int bi = 0;
    for (int v = 0; v < VV; ++v) {
      float x = Lt[tid][v] + out_b[v];
      if (x > best) { best = x; bi = v; }      // strict > keeps first max (argmax)
    }
    int eo = eos[b];
    x_out[(size_t)b * MAXLEN + t] = eo ? 0 : bi;
    if (!eo && bi == 2) { eos[b] = 1; end_pads[b] = t + 1; }
    tok[b] = bi;
  }
}

// ---------------------------------------------------------------------------
// Host launcher
// ---------------------------------------------------------------------------
extern "C" void kernel_launch(void* const* d_in, const int* in_sizes, int n_in,
                              void* d_out, int out_size, void* d_ws, size_t ws_size,
                              hipStream_t stream) {
  const float* z     = (const float*)d_in[0];
  const float* emb   = (const float*)d_in[1];
  const float* lat_W = (const float*)d_in[2];
  const float* lat_b = (const float*)d_in[3];
  const float* Wih0  = (const float*)d_in[4];
  const float* Whh0  = (const float*)d_in[5];
  const float* bih0  = (const float*)d_in[6];
  const float* bhh0  = (const float*)d_in[7];
  const float* Wih1  = (const float*)d_in[8];
  const float* Whh1  = (const float*)d_in[9];
  const float* bih1  = (const float*)d_in[10];
  const float* bhh1  = (const float*)d_in[11];
  const float* out_W = (const float*)d_in[12];
  const float* out_b = (const float*)d_in[13];

  int* x_out    = (int*)d_out;             // [B, MAXLEN]
  int* end_pads = x_out + (size_t)BB * MAXLEN;

  char* wsp = (char*)d_ws;
  auto alloc = [&](size_t bytes) -> void* {
    void* p = (void*)wsp;
    wsp += (bytes + 255) & ~(size_t)255;
    return p;
  };
  u16t* zbf    = (u16t*)alloc((size_t)BB * DD * 2);
  u16t* embbf  = (u16t*)alloc((size_t)VV * HH * 2);
  u16t* owbf   = (u16t*)alloc((size_t)VV * HH * 2);
  u16t* latwbf = (u16t*)alloc((size_t)HH * DD * 2);
  u16t* Wp0    = (u16t*)alloc((size_t)2048 * 1536 * 2);
  u16t* Wp1    = (u16t*)alloc((size_t)2048 * 1024 * 2);
  float* bp0   = (float*)alloc(2048 * 4);
  float* bp1   = (float*)alloc(2048 * 4);
  float* h0f[2] = { (float*)alloc((size_t)BB * HH * 4), (float*)alloc((size_t)BB * HH * 4) };
  float* h1f[2] = { (float*)alloc((size_t)BB * HH * 4), (float*)alloc((size_t)BB * HH * 4) };
  u16t*  h0b[2] = { (u16t*)alloc((size_t)BB * HH * 2),  (u16t*)alloc((size_t)BB * HH * 2) };
  u16t*  h1b[2] = { (u16t*)alloc((size_t)BB * HH * 2),  (u16t*)alloc((size_t)BB * HH * 2) };
  int* tokb = (int*)alloc(BB * 4);
  int* eosb = (int*)alloc(BB * 4);

  auto cvt = [&](const float* s, u16t* d, int n) {
    k_f2bf<<<(n + 255) / 256, 256, 0, stream>>>(s, d, n);
  };
  cvt(z, zbf, BB * DD);
  cvt(emb, embbf, VV * HH);
  cvt(out_W, owbf, VV * HH);
  cvt(lat_W, latwbf, HH * DD);
  k_pack_gates<<<2048, 256, 0, stream>>>(Wih0, Whh0, bih0, bhh0, Wp0, bp0, HH + DD);
  k_pack_gates<<<2048, 256, 0, stream>>>(Wih1, Whh1, bih1, bhh1, Wp1, bp1, HH);
  k_init_seq<<<(BB + 255) / 256, 256, 0, stream>>>(tokb, eosb, x_out, end_pads);

  // h0 = z @ lat_W^T + lat_b  (broadcast to both GRU layers, buffer 0)
  k_gru_gemm<<<dim3(HH / 64, BB / 128), 256, 0, stream>>>(
      zbf, nullptr, nullptr, nullptr, nullptr,
      latwbf, 512, lat_b, nullptr,
      h0f[0], h0b[0], h1f[0], h1b[0], /*mode=*/1);

  // autoregressive decode: 99 sequential steps
  for (int t = 1; t < MAXLEN; ++t) {
    int cur = (t - 1) & 1, nxt = cur ^ 1;
    // layer 0: A = [emb[tok] | z | h0_cur], K = 1536
    k_gru_gemm<<<dim3(32, BB / 128), 256, 0, stream>>>(
        nullptr, zbf, h0b[cur], tokb, embbf,
        Wp0, 1536, bp0, h0f[cur],
        h0f[nxt], h0b[nxt], nullptr, nullptr, /*mode=*/0);
    // layer 1: A = [h0_new | h1_cur], K = 1024
    k_gru_gemm<<<dim3(32, BB / 128), 256, 0, stream>>>(
        h0b[nxt], h1b[cur], nullptr, nullptr, nullptr,
        Wp1, 1024, bp1, h1f[cur],
        h1f[nxt], h1b[nxt], nullptr, nullptr, /*mode=*/0);
    // logits + argmax + token/eos bookkeeping
    k_logits_argmax<<<BB / 16, 256, 0, stream>>>(
        h1b[nxt], owbf, out_b, tokb, eosb, x_out, end_pads, t);
  }
  (void)in_sizes; (void)n_in; (void)out_size; (void)ws_size;
}
